// TriangleUpdate_68178310857256
// MI455X (gfx1250) — matrix operand order repaired and checked by reading, not compiled, not run on hardware
//
#include <hip/hip_runtime.h>

#define EW 64
#define NH 8
#define DD 8
#define NB 2
#define NN 384
#define LN_EPS 1e-5f

typedef __attribute__((ext_vector_type(16))) _Float16 v16h;
typedef __attribute__((ext_vector_type(8)))  float    v8f;

union Frag { v16h v; _Float16 h[16]; unsigned u[8]; };

// ---------------------------------------------------------------------------
// Kernel 0: pack weights into fp16, column-major ([n][k]) so WMMA B-fragments
// load as contiguous dwords.
// ---------------------------------------------------------------------------
__global__ void tri_prep_kernel(const float* __restrict__ W_V,
                                const float* __restrict__ W_EG,
                                const float* __restrict__ W_O,
                                const float* __restrict__ b_V,
                                const float* __restrict__ b_EG,
                                _Float16* __restrict__ Wcat_t,   // [160][64]
                                float* __restrict__ bcat,        // [160]
                                _Float16* __restrict__ WOt) {    // [64][128]
  const int stride = gridDim.x * blockDim.x;
  const int t0 = blockIdx.x * blockDim.x + threadIdx.x;
  for (int idx = t0; idx < 160 * 64; idx += stride) {
    int n = idx >> 6, k = idx & 63;
    float w = (n < 128) ? W_V[k * 128 + n] : W_EG[k * 32 + (n - 128)];
    Wcat_t[idx] = (_Float16)w;
  }
  for (int idx = t0; idx < 160; idx += stride)
    bcat[idx] = (idx < 128) ? b_V[idx] : b_EG[idx - 128];
  for (int idx = t0; idx < 64 * 128; idx += stride) {
    int n = idx >> 7, k = idx & 127;
    WOt[idx] = (_Float16)W_O[k * 64 + n];
  }
}

// ---------------------------------------------------------------------------
// Kernel 1: LayerNorm + fused V/EG projections via WMMA.
// One wave handles 16 edges (fixed b,row; 16 consecutive cols).
// A = e_ln [16x64] fp16 (staged in LDS), B = Wcat_t [64x160].
// ---------------------------------------------------------------------------
__global__ __launch_bounds__(128)
void tri_ln_proj_kernel(const float* __restrict__ e,
                        const float* __restrict__ ln_w,
                        const float* __restrict__ ln_b,
                        const _Float16* __restrict__ Wcat_t,
                        const float* __restrict__ bcat,
                        _Float16* __restrict__ Vin_t,   // [16][3072][384]
                        _Float16* __restrict__ Vout_t,  // [16][3072][384]
                        float* __restrict__ Ein,        // [16][384][384]
                        float* __restrict__ Gin,
                        float* __restrict__ Eout,
                        float* __restrict__ Gout) {
  __shared__ _Float16 lds[4][16][EW];   // 8 KB
  const int wave = threadIdx.x >> 5;
  const int lane = threadIdx.x & 31;
  const long tile = (long)blockIdx.x * 4 + wave;   // 16-edge tile
  const long e0   = tile * 16;
  const int  ce0  = (int)(e0 % NN);
  const long br   = e0 / NN;            // b*NN + re
  const int  re   = (int)(br % NN);
  const int  b    = (int)(br / NN);

  // ---- LayerNorm: 2 lanes per edge, 32 features each ----
  {
    const int m  = lane >> 1;
    const int fh = (lane & 1) * 32;
    const float* ep = e + ((br * NN + ce0 + m) << 6) + fh;
    float x[32];
    float s = 0.f, ss = 0.f;
#pragma unroll
    for (int f = 0; f < 32; ++f) { x[f] = ep[f]; s += x[f]; ss += x[f] * x[f]; }
    s  += __shfl_xor(s, 1, 32);
    ss += __shfl_xor(ss, 1, 32);
    float mu  = s * (1.f / 64.f);
    float var = ss * (1.f / 64.f) - mu * mu;
    float inv = rsqrtf(var + LN_EPS);
#pragma unroll
    for (int f = 0; f < 32; ++f) {
      float v = (x[f] - mu) * inv * ln_w[fh + f] + ln_b[fh + f];
      lds[wave][m][fh + f] = (_Float16)v;
    }
  }
  __syncthreads();

  const int m  = lane & 15;
  const int hi = lane >> 4;

  // A-fragments for both 32-wide K chunks of the 64-deep contraction
  Frag af0, af1;
#pragma unroll
  for (int i = 0; i < 8; ++i) {
    int kb = ((i >> 2) << 4) + (hi << 3) + ((i & 3) << 1);
    af0.u[i] = *(const unsigned*)&lds[wave][m][kb];
    af1.u[i] = *(const unsigned*)&lds[wave][m][kb + 32];
  }

  for (int t = 0; t < 10; ++t) {          // 10 tiles of 16 output cols = 160
    Frag bf0, bf1;
    const _Float16* wb = Wcat_t + (t * 16 + m) * EW;
#pragma unroll
    for (int i = 0; i < 8; ++i) {
      int kk = ((i >> 2) << 4) + (hi << 3) + ((i & 3) << 1);
      bf0.u[i] = *(const unsigned*)(wb + kk);
      bf1.u[i] = *(const unsigned*)(wb + kk + 32);
    }
    v8f acc = {};
    acc = __builtin_amdgcn_wmma_f32_16x16x32_f16(false, af0.v, false, bf0.v,
                                                 (short)0, acc, false, false);
    acc = __builtin_amdgcn_wmma_f32_16x16x32_f16(false, af1.v, false, bf1.v,
                                                 (short)0, acc, false, false);
    const int col  = t * 16 + m;
    const float bias = bcat[col];
#pragma unroll
    for (int r = 0; r < 8; ++r) {
      const int row = r + (hi << 3);
      const int ce  = ce0 + row;
      const float val = acc[r] + bias;
      if (col < 64) {                       // V_in: j=re, k=ce
        int d = col >> 3, h = col & 7;
        Vin_t[(((long)(b * NH + h) * (NN * DD)) + (long)re * DD + d) * NN + ce] =
            (_Float16)val;
      } else if (col < 128) {               // V_out: k=re, j=ce
        int v = col - 64, d = v >> 3, h = v & 7;
        Vout_t[(((long)(b * NH + h) * (NN * DD)) + (long)ce * DD + d) * NN + re] =
            (_Float16)val;
      } else {
        int g = col - 128, h = g & 7, sel = g >> 3;
        long idxin  = ((long)(b * NH + h) * NN + re) * NN + ce;
        long idxout = ((long)(b * NH + h) * NN + ce) * NN + re;
        if      (sel == 0) Ein[idxin]   = val;
        else if (sel == 1) Gin[idxin]   = val;
        else if (sel == 2) Eout[idxout] = val;
        else               Gout[idxout] = val;
      }
    }
  }
}

// ---------------------------------------------------------------------------
// Kernel 2: softmax over k (+mask) * sigmoid gate -> fp16 A matrices.
// blockIdx: x = i (row), y = bh, z = io (0=in, 1=out). 128 threads, 384 cols.
// ---------------------------------------------------------------------------
__global__ __launch_bounds__(128)
void tri_softmax_gate_kernel(const float* __restrict__ Ein,
                             const float* __restrict__ Gin,
                             const float* __restrict__ Eout,
                             const float* __restrict__ Gout,
                             const float* __restrict__ mask,
                             _Float16* __restrict__ Ain,
                             _Float16* __restrict__ Aout) {
  const int i  = blockIdx.x;
  const int bh = blockIdx.y;
  const int io = blockIdx.z;
  const int b  = bh >> 3;
  const int t  = threadIdx.x;
  const long row = ((long)bh * NN + i) * NN;
  const float* E = (io ? Eout : Ein) + row;
  const float* G = (io ? Gout : Gin) + row;
  _Float16*    A = (io ? Aout : Ain) + row;

  __shared__ float red[128];
  float vals[3], ms[3];
  float mx = -1e30f;
#pragma unroll
  for (int c = 0; c < 3; ++c) {
    int k = t + c * 128;
    float mk = io ? mask[((long)b * NN + k) * NN + i]
                  : mask[((long)b * NN + i) * NN + k];
    ms[c] = mk;
    vals[c] = E[k] + mk;
    mx = fmaxf(mx, vals[c]);
  }
  red[t] = mx; __syncthreads();
  for (int s = 64; s > 0; s >>= 1) {
    if (t < s) red[t] = fmaxf(red[t], red[t + s]);
    __syncthreads();
  }
  mx = red[0]; __syncthreads();

  float sum = 0.f;
#pragma unroll
  for (int c = 0; c < 3; ++c) { vals[c] = __expf(vals[c] - mx); sum += vals[c]; }
  red[t] = sum; __syncthreads();
  for (int s = 64; s > 0; s >>= 1) {
    if (t < s) red[t] += red[t + s];
    __syncthreads();
  }
  const float inv = 1.f / red[0];
#pragma unroll
  for (int c = 0; c < 3; ++c) {
    int k = t + c * 128;
    float g = 1.f / (1.f + __expf(-(G[k] + ms[c])));
    A[k] = (_Float16)(vals[c] * inv * g);
  }
}

// ---------------------------------------------------------------------------
// Kernel 3: the big einsums as 32 GEMMs: C[384x3072] = A[384x384] * Vt^T.
// One wave computes a 32x64 tile (2 A-frags reused over 4 B-frags -> 8 WMMAs
// per K-step), 12 K-steps of 32. Next K-step is software-prefetched
// (global_prefetch_b8). Output scattered into Va[b][i][j][c] fp16 with
// c = (io?64:0) + d*8 + h.
// ---------------------------------------------------------------------------
__global__ __launch_bounds__(32)
void tri_attn_gemm_kernel(const _Float16* __restrict__ Ain,
                          const _Float16* __restrict__ Aout,
                          const _Float16* __restrict__ Vin_t,
                          const _Float16* __restrict__ Vout_t,
                          _Float16* __restrict__ Va) {
  const int ntile = blockIdx.x;          // 0..47 (64 cols each)
  const int mtile = blockIdx.y;          // 0..11 (32 rows each)
  const int gz    = blockIdx.z;          // io*16 + bh
  const int io    = gz >> 4;
  const int bh    = gz & 15;
  const int b     = bh >> 3, h = bh & 7;
  const int lane  = threadIdx.x;
  const int m     = lane & 15, hi = lane >> 4;

  const _Float16* A0 =
      (io ? Aout : Ain) + ((long)bh * NN + mtile * 32 + m) * NN;
  const _Float16* A1 = A0 + 16 * NN;
  const _Float16* Vt = (io ? Vout_t : Vin_t) + (long)bh * (NN * DD) * NN;
  const _Float16* Bp[4];
#pragma unroll
  for (int nt = 0; nt < 4; ++nt)
    Bp[nt] = Vt + (long)(ntile * 64 + nt * 16 + m) * NN;

  v8f acc[8];
#pragma unroll
  for (int q = 0; q < 8; ++q) acc[q] = (v8f){};

  for (int ks = 0; ks < NN; ks += 32) {
    // prefetch next K-step while this one computes
    const int ksn = ks + 32;
    if (ksn < NN) {
      __builtin_prefetch(A0 + ksn, 0, 3);
      __builtin_prefetch(A1 + ksn, 0, 3);
#pragma unroll
      for (int nt = 0; nt < 4; ++nt) __builtin_prefetch(Bp[nt] + ksn, 0, 3);
    }
    Frag af0, af1;
#pragma unroll
    for (int i = 0; i < 8; ++i) {
      int kk = ks + ((i >> 2) << 4) + (hi << 3) + ((i & 3) << 1);
      af0.u[i] = *(const unsigned*)(A0 + kk);
      af1.u[i] = *(const unsigned*)(A1 + kk);
    }
#pragma unroll
    for (int nt = 0; nt < 4; ++nt) {
      Frag bf;
#pragma unroll
      for (int i = 0; i < 8; ++i)
        bf.u[i] = *(const unsigned*)(Bp[nt] + ks + ((i >> 2) << 4) + (hi << 3) + ((i & 3) << 1));
      acc[nt] = __builtin_amdgcn_wmma_f32_16x16x32_f16(
          false, af0.v, false, bf.v, (short)0, acc[nt], false, false);
      acc[4 + nt] = __builtin_amdgcn_wmma_f32_16x16x32_f16(
          false, af1.v, false, bf.v, (short)0, acc[4 + nt], false, false);
    }
  }
#pragma unroll
  for (int mi = 0; mi < 2; ++mi) {
#pragma unroll
    for (int nt = 0; nt < 4; ++nt) {
      v8f a = acc[mi * 4 + nt];
      const int np = ntile * 64 + nt * 16 + m;   // (j,d) column
      const int j = np >> 3, d = np & 7;
      const int cch = (io ? 64 : 0) + d * 8 + h;
#pragma unroll
      for (int r = 0; r < 8; ++r) {
        const int ig = mtile * 32 + mi * 16 + r + (hi << 3);
        Va[(((long)b * NN + ig) * NN + j) * 128 + cch] = (_Float16)a[r];
      }
    }
  }
}

// ---------------------------------------------------------------------------
// Kernel 4: output projection out = Va[.,128] @ W_O[128,64] + b_O (fp32 out).
// One wave per 16 edges; 4 K-steps, 4 N-tiles -> 16 WMMAs.
// ---------------------------------------------------------------------------
__global__ __launch_bounds__(128)
void tri_out_proj_kernel(const _Float16* __restrict__ Va,
                         const _Float16* __restrict__ WOt,   // [64][128]
                         const float* __restrict__ b_O,
                         float* __restrict__ out) {
  const int wave = threadIdx.x >> 5;
  const int lane = threadIdx.x & 31;
  const long e0  = ((long)blockIdx.x * 4 + wave) * 16;
  const int m = lane & 15, hi = lane >> 4;

  const _Float16* Ap = Va + (e0 + m) * 128;
  v8f acc0 = {}, acc1 = {}, acc2 = {}, acc3 = {};
  for (int ks = 0; ks < 128; ks += 32) {
    if (ks + 32 < 128) __builtin_prefetch(Ap + ks + 32, 0, 3);
    Frag af;
#pragma unroll
    for (int i = 0; i < 8; ++i) {
      int kk = ks + ((i >> 2) << 4) + (hi << 3) + ((i & 3) << 1);
      af.u[i] = *(const unsigned*)(Ap + kk);
    }
#pragma unroll
    for (int nt = 0; nt < 4; ++nt) {
      Frag bf;
      const _Float16* Bp = WOt + (nt * 16 + m) * 128 + ks;
#pragma unroll
      for (int i = 0; i < 8; ++i)
        bf.u[i] = *(const unsigned*)(Bp + ((i >> 2) << 4) + (hi << 3) + ((i & 3) << 1));
      v8f& acc = nt == 0 ? acc0 : nt == 1 ? acc1 : nt == 2 ? acc2 : acc3;
      acc = __builtin_amdgcn_wmma_f32_16x16x32_f16(false, af.v, false, bf.v,
                                                   (short)0, acc, false, false);
    }
  }
#pragma unroll
  for (int nt = 0; nt < 4; ++nt) {
    v8f acc = nt == 0 ? acc0 : nt == 1 ? acc1 : nt == 2 ? acc2 : acc3;
    const int col = nt * 16 + m;
    const float bias = b_O[col];
#pragma unroll
    for (int r = 0; r < 8; ++r) {
      const long eidx = e0 + r + (hi << 3);
      out[eidx * 64 + col] = acc[r] + bias;
    }
  }
}

// ---------------------------------------------------------------------------
extern "C" void kernel_launch(void* const* d_in, const int* in_sizes, int n_in,
                              void* d_out, int out_size, void* d_ws, size_t ws_size,
                              hipStream_t stream) {
  (void)in_sizes; (void)n_in; (void)out_size; (void)ws_size;
  const float* e    = (const float*)d_in[0];
  const float* mask = (const float*)d_in[1];
  const float* ln_w = (const float*)d_in[2];
  const float* ln_b = (const float*)d_in[3];
  const float* W_V  = (const float*)d_in[4];
  const float* b_V  = (const float*)d_in[5];
  const float* W_EG = (const float*)d_in[6];
  const float* b_EG = (const float*)d_in[7];
  const float* W_O  = (const float*)d_in[8];
  const float* b_O  = (const float*)d_in[9];
  float* out = (float*)d_out;

  char* ws = (char*)d_ws;
  size_t off = 0;
  auto alloc = [&](size_t bytes) -> char* {
    char* p = ws + off;
    off = (off + bytes + 255) & ~(size_t)255;
    return p;
  };
  _Float16* Wcat_t = (_Float16*)alloc(160 * 64 * 2);
  float*    bcat   = (float*)   alloc(160 * 4);
  _Float16* WOt    = (_Float16*)alloc(64 * 128 * 2);
  float* Ein  = (float*)alloc((size_t)16 * NN * NN * 4);
  float* Gin  = (float*)alloc((size_t)16 * NN * NN * 4);
  float* Eout = (float*)alloc((size_t)16 * NN * NN * 4);
  float* Gout = (float*)alloc((size_t)16 * NN * NN * 4);
  _Float16* Ain  = (_Float16*)alloc((size_t)16 * NN * NN * 2);
  _Float16* Aout = (_Float16*)alloc((size_t)16 * NN * NN * 2);
  _Float16* Vin_t  = (_Float16*)alloc((size_t)16 * NN * DD * NN * 2);
  _Float16* Vout_t = (_Float16*)alloc((size_t)16 * NN * DD * NN * 2);
  _Float16* Va = (_Float16*)alloc((size_t)NB * NN * NN * 128 * 2);

  tri_prep_kernel<<<64, 256, 0, stream>>>(W_V, W_EG, W_O, b_V, b_EG,
                                          Wcat_t, bcat, WOt);
  tri_ln_proj_kernel<<<4608, 128, 0, stream>>>(e, ln_w, ln_b, Wcat_t, bcat,
                                               Vin_t, Vout_t, Ein, Gin, Eout, Gout);
  tri_softmax_gate_kernel<<<dim3(NN, 16, 2), 128, 0, stream>>>(
      Ein, Gin, Eout, Gout, mask, Ain, Aout);
  tri_attn_gemm_kernel<<<dim3(48, 12, 32), 32, 0, stream>>>(
      Ain, Aout, Vin_t, Vout_t, Va);
  tri_out_proj_kernel<<<4608, 128, 0, stream>>>(Va, WOt, b_O, out);
}
